// TFEsmSelfAttention_55327768707660
// MI455X (gfx1250) — matrix operand order repaired and checked by reading, not compiled, not run on hardware
//
#include <hip/hip_runtime.h>
#include <hip/hip_bf16.h>

// Problem constants (from reference)
#define Bv   2
#define Sv   2048
#define HIDv 1280
#define Hv   20
#define Dv   64
#define Mv   (Bv * Sv)        // 4096 rows of X
#define KT40 (HIDv / 32)      // 40 k-tiles of 32

typedef _Float16 h8  __attribute__((ext_vector_type(8)));
typedef _Float16 h16 __attribute__((ext_vector_type(16)));
typedef float    f8  __attribute__((ext_vector_type(8)));

__device__ __forceinline__ f8 zero8() {
  f8 z;
#pragma unroll
  for (int i = 0; i < 8; ++i) z[i] = 0.0f;
  return z;
}

__device__ __forceinline__ h16 cat16(h8 lo, h8 hi) {
  return __builtin_shufflevector(lo, hi, 0,1,2,3,4,5,6,7,8,9,10,11,12,13,14,15);
}

// 16 contiguous halves -> v16h (two 128-bit loads; global or LDS)
__device__ __forceinline__ h16 ld16c(const _Float16* p) {
  h8 a = *(const h8*)(p);
  h8 b = *(const h8*)(p + 8);
  return cat16(a, b);
}

__device__ __forceinline__ f8 wmma16(h16 a, h16 b, f8 c) {
  // D = A(16x32 f16) * B(32x16 f16) + C(16x16 f32)
  return __builtin_amdgcn_wmma_f32_16x16x32_f16(false, a, false, b, (short)0, c,
                                                false, false);
}

// Async global -> LDS copy (ASYNCcnt-tracked), 16 bytes per lane.
__device__ __forceinline__ void async_ld_b128(const void* gptr, void* lptr) {
  unsigned lds = (unsigned)(uintptr_t)lptr;            // addr[31:0] = LDS offset
  unsigned long long ga = (unsigned long long)(uintptr_t)gptr;
  asm volatile("global_load_async_to_lds_b128 %0, %1, off"
               :: "v"(lds), "v"(ga) : "memory");
}

// ---------------------------------------------------------------------------
// Kernel 0a: fp32 -> f16 cast of hidden states
// ---------------------------------------------------------------------------
__global__ __launch_bounds__(256) void cast_x_kernel(const float* __restrict__ X,
                                                     _Float16* __restrict__ Xh,
                                                     int n) {
  int i = blockIdx.x * blockDim.x + threadIdx.x;
  if (i < n) Xh[i] = (_Float16)X[i];
}

// ---------------------------------------------------------------------------
// Kernel 0b: pack Wq/Wk/Wv (fp32 [K=HID][N=HID] row-major) into f16
// B-fragment lane order:
//   pos = ((n_tile*KT40 + k_tile)*32 + lane)*16 + e
//   lane = (n%16) + 16*((k%32) >= 16),  e = k%16
// ---------------------------------------------------------------------------
__global__ __launch_bounds__(256) void pack_w_kernel(const float* __restrict__ Wq,
                                                     const float* __restrict__ Wk,
                                                     const float* __restrict__ Wv,
                                                     _Float16* __restrict__ Wp) {
  int i = blockIdx.x * blockDim.x + threadIdx.x;
  if (i >= 3 * HIDv * HIDv) return;
  int which = i / (HIDv * HIDv);
  int rem   = i - which * (HIDv * HIDv);
  int k = rem / HIDv;
  int n = rem - k * HIDv;
  const float* W = (which == 0) ? Wq : (which == 1) ? Wk : Wv;
  int lane = (n & 15) + (((k & 31) >> 4) << 4);
  int e    = k & 15;
  int nt = n >> 4, kt = k >> 5;
  size_t pos = (size_t)which * HIDv * HIDv +
               ((((size_t)nt * KT40 + kt) * 32 + lane) * 16 + e);
  Wp[pos] = (_Float16)W[(size_t)k * HIDv + n];
}

// ---------------------------------------------------------------------------
// Kernel 1: QKV projection (WMMA f16) + bias + q-scale + RoPE, storing
//   Q,K -> f16 [B,H,S,D] row-major, V -> f16 [B,H,D,S] (transposed).
// One wave handles 16 rows x 64 cols (one head) of one of {q,k,v}.
// ---------------------------------------------------------------------------
__global__ __launch_bounds__(128) void qkv_rope_kernel(
    const _Float16* __restrict__ Xh,     // [Mv][HIDv]
    const _Float16* __restrict__ Wp,     // packed, 3 * HID*HID halves
    const float* __restrict__ bq, const float* __restrict__ bk,
    const float* __restrict__ bv,
    _Float16* __restrict__ Qh, _Float16* __restrict__ Kh,
    _Float16* __restrict__ Vt) {
  int wave = (blockIdx.x * blockDim.x + threadIdx.x) >> 5;
  int lane = threadIdx.x & 31;
  if (wave >= 256 * Hv * 3) return;
  int m_t  = wave & 255;          // 256 row tiles of 16
  int rest = wave >> 8;           // 0..59
  int head  = rest % Hv;
  int which = rest / Hv;          // 0=q 1=k 2=v

  int lm = lane & 15;
  int lhi = (lane >= 16) ? 1 : 0;
  int aoff = lhi ? 8 : 0;

  const _Float16* xrow  = Xh + (size_t)(m_t * 16 + lm) * HIDv;
  const _Float16* wbase = Wp + (size_t)which * HIDv * HIDv;

  f8 acc[4];
#pragma unroll
  for (int t = 0; t < 4; ++t) acc[t] = zero8();

  for (int kt = 0; kt < KT40; ++kt) {
    int kb = kt * 32;
    __builtin_prefetch(xrow + kb + 128, 0, 1);  // global_prefetch_b8
    // A fragment: row lm; element e -> k=kb+2*aoff+e (contiguous)
    h16 A = ld16c(xrow + kb + 2 * aoff);
#pragma unroll
    for (int t = 0; t < 4; ++t) {
      int nt = head * 4 + t;
      const _Float16* bp =
          wbase + ((((size_t)nt * KT40 + kt) * 32 + lane) * 16);
      h16 Bf = ld16c(bp);
      acc[t] = wmma16(A, Bf, acc[t]);
    }
  }

  // bias (per output column)
  const float* bias = (which == 0) ? bq : (which == 1) ? bk : bv;
#pragma unroll
  for (int t = 0; t < 4; ++t) {
    float bc = bias[head * Dv + t * 16 + lm];
#pragma unroll
    for (int r = 0; r < 8; ++r) acc[t][r] += bc;
  }

  int rbase = m_t * 16 + lhi * 8;  // first output row for this lane's VGPR0

  if (which == 2) {
    // V -> transposed [B,H,D,S]
#pragma unroll
    for (int t = 0; t < 4; ++t) {
      int d = t * 16 + lm;
#pragma unroll
      for (int r = 0; r < 8; ++r) {
        int orow = rbase + r;
        int bb = orow >> 11, ss = orow & (Sv - 1);
        Vt[(((size_t)bb * Hv + head) * Dv + d) * Sv + ss] = (_Float16)acc[t][r];
      }
    }
  } else {
    // Q (scaled) or K, with RoPE. Column pair (c, c+32) lives in acc[t], acc[t+2].
    float qs = (which == 0) ? 0.125f : 1.0f;  // D^-0.5 = 1/8
    _Float16* dst = (which == 0) ? Qh : Kh;
#pragma unroll
    for (int r = 0; r < 8; ++r) {
      int orow = rbase + r;
      int bb = orow >> 11, ss = orow & (Sv - 1);
      _Float16* op = dst + (((size_t)bb * Hv + head) * Sv + ss) * Dv;
#pragma unroll
      for (int t = 0; t < 2; ++t) {
        int c = t * 16 + lm;  // [0,32)
        float x0 = acc[t][r] * qs;
        float x1 = acc[t + 2][r] * qs;
        // inv_freq = 10000^(-c/32) = exp(-c * ln(1e4)/32)
        float fr = (float)ss * __expf((float)c * -0.28782313662425572f);
        float sn, cs;
        __sincosf(fr, &sn, &cs);
        op[c]      = (_Float16)(x0 * cs - x1 * sn);
        op[c + 32] = (_Float16)(x1 * cs + x0 * sn);
      }
    }
  }
}

// ---------------------------------------------------------------------------
// Kernel 2: flash attention. 128 threads = 4 waves per block; each wave owns
// 16 queries (64 per block). K (32x64) and V^T (64x32) tiles are staged into
// LDS with double-buffered async global->LDS copies shared by all 4 waves.
// ---------------------------------------------------------------------------
__global__ __launch_bounds__(128) void flash_attn_kernel(
    const _Float16* __restrict__ Qh,  // [B,H,S,D]
    const _Float16* __restrict__ Kh,  // [B,H,S,D]
    const _Float16* __restrict__ Vt,  // [B,H,D,S]
    const float* __restrict__ mask,   // [B,1,1,S]
    float* __restrict__ out) {        // [B,S,H*D]
  __shared__ _Float16 Kst[2][32 * 64];  // [buf][key][d]      8 KB
  __shared__ _Float16 Vst[2][64 * 32];  // [buf][d][key]      8 KB
  __shared__ _Float16 Pst[4][16 * 32];  // per-wave P tile    4 KB

  int tid = threadIdx.x;
  int w = tid >> 5;          // wave 0..3
  int lane = tid & 31;
  int lm = lane & 15;
  int lhi = (lane >= 16) ? 1 : 0;
  int aoff = lhi ? 8 : 0;    // A-fragment half shift
  int boff = lhi ? 16 : 0;   // B-fragment half shift

  int blk = blockIdx.x;
  int qchunk = blk & 31;     // S/64 = 32 query chunks
  int bh = blk >> 5;         // 0..39
  int b = bh / Hv, h = bh % Hv;
  int qrow0 = qchunk * 64 + w * 16;  // this wave's first query row

  const _Float16* kbase = Kh + (size_t)bh * Sv * Dv;
  const _Float16* vbase = Vt + (size_t)bh * Dv * Sv;
  const float*    mrow  = mask + (size_t)b * Sv;

  // Q A-fragments (registers for whole key loop): db = 0, 32
  const _Float16* q0 =
      Qh + ((size_t)bh * Sv + qrow0) * Dv + (size_t)lm * Dv + 2 * aoff;
  h16 aq0 = ld16c(q0);
  h16 aq1 = ld16c(q0 + 32);

  f8 acc_o[4];
#pragma unroll
  for (int t = 0; t < 4; ++t) acc_o[t] = zero8();
  float mrun[8], lrun[8];
#pragma unroll
  for (int r = 0; r < 8; ++r) { mrun[r] = -1e30f; lrun[r] = 0.0f; }

  const int NT = Sv / 32;  // 64 key tiles

  // Issue async K/V tile loads: 128 threads x (2 K + 2 V) x 16B = 8 KB/tile.
  auto issue_tile = [&](int tile, int buf) {
    int kb = tile * 32;
#pragma unroll
    for (int j = 0; j < 2; ++j) {
      int c = tid + j * 128;                 // chunk id 0..255
      int krow = c >> 3, koff = (c & 7) * 8; // K: 8 chunks per 64-half row
      async_ld_b128(kbase + (size_t)(kb + krow) * Dv + koff,
                    &Kst[buf][krow * 64 + koff]);
      int vrow = c >> 2, voff = (c & 3) * 8; // V: 4 chunks per 32-half row
      async_ld_b128(vbase + (size_t)vrow * Sv + kb + voff,
                    &Vst[buf][vrow * 32 + voff]);
    }
  };

  issue_tile(0, 0);
  issue_tile(1, 1);

  for (int i = 0; i < NT; ++i) {
    int buf = i & 1;
    int kb = i * 32;
    // Retire this tile's async copies (next tile's 4/wave may stay in flight).
    if (i + 1 < NT)
      asm volatile("s_wait_asynccnt 0x4" ::: "memory");
    else
      asm volatile("s_wait_asynccnt 0x0" ::: "memory");
    __syncthreads();  // all waves' chunks of tile i are in LDS

    const _Float16* kst = &Kst[buf][0];
    const _Float16* vst = &Vst[buf][0];
    _Float16* pst = &Pst[w][0];

    // ---- scores: two 16x16 tiles over this 32-key block (K from LDS) ----
    f8 sc[2];
#pragma unroll
    for (int kt2 = 0; kt2 < 2; ++kt2) {
      const _Float16* kr = kst + (kt2 * 16 + lm) * 64 + boff;
      h16 bk0 = ld16c(kr);       // kd = 0..31 slice
      h16 bk1 = ld16c(kr + 32);  // kd = 32..63 slice
      f8 s = zero8();
      s = wmma16(aq0, bk0, s);
      s = wmma16(aq1, bk1, s);
      float mv = mrow[kb + kt2 * 16 + lm];
#pragma unroll
      for (int r = 0; r < 8; ++r) s[r] += mv;
      sc[kt2] = s;
    }

    // ---- online softmax (each 16-lane half owns 8 rows x 32 cols) ----
#pragma unroll
    for (int r = 0; r < 8; ++r) {
      float v = fmaxf(sc[0][r], sc[1][r]);
#pragma unroll
      for (int off = 1; off < 16; off <<= 1)
        v = fmaxf(v, __shfl_xor(v, off, 16));
      float mnew = fmaxf(mrun[r], v);
      float p0 = __expf(sc[0][r] - mnew);
      float p1 = __expf(sc[1][r] - mnew);
      int prow = r + lhi * 8;  // C-layout row this lane holds
      pst[prow * 32 + lm]      = (_Float16)p0;
      pst[prow * 32 + 16 + lm] = (_Float16)p1;
      float sum = p0 + p1;
#pragma unroll
      for (int off = 1; off < 16; off <<= 1)
        sum += __shfl_xor(sum, off, 16);
      float scale = __expf(mrun[r] - mnew);
      lrun[r] = lrun[r] * scale + sum;
      mrun[r] = mnew;
#pragma unroll
      for (int t = 0; t < 4; ++t) acc_o[t][r] *= scale;
    }
    __syncthreads();  // P tiles visible (and uniform across waves)

    // ---- P (16x32, LDS A fragment) @ V (32x64, LDS B fragments) ----
    const _Float16* pr = pst + lm * 32 + boff;
    h16 ap = ld16c(pr);
#pragma unroll
    for (int t = 0; t < 4; ++t) {
      const _Float16* vr = vst + (t * 16 + lm) * 32 + boff;
      h16 bvf = ld16c(vr);
      acc_o[t] = wmma16(ap, bvf, acc_o[t]);
    }
    __syncthreads();  // everyone done reading buf before it is overwritten

    if (i + 2 < NT) issue_tile(i + 2, buf);
  }

  // ---- normalize and write fp32 output [B,S,H*D] ----
  int srow0 = qrow0 + lhi * 8;
#pragma unroll
  for (int t = 0; t < 4; ++t) {
    int d = t * 16 + lm;
#pragma unroll
    for (int r = 0; r < 8; ++r) {
      int s = srow0 + r;
      out[((size_t)b * Sv + s) * HIDv + h * Dv + d] = acc_o[t][r] / lrun[r];
    }
  }
}

// ---------------------------------------------------------------------------
extern "C" void kernel_launch(void* const* d_in, const int* in_sizes, int n_in,
                              void* d_out, int out_size, void* d_ws,
                              size_t ws_size, hipStream_t stream) {
  const float* X    = (const float*)d_in[0];
  const float* mask = (const float*)d_in[1];
  const float* Wq   = (const float*)d_in[2];
  const float* bq   = (const float*)d_in[3];
  const float* Wk   = (const float*)d_in[4];
  const float* bk   = (const float*)d_in[5];
  const float* Wv   = (const float*)d_in[6];
  const float* bv   = (const float*)d_in[7];
  float* out = (float*)d_out;

  char* ws = (char*)d_ws;
  const size_t nX  = (size_t)Mv * HIDv;            // 5,242,880 halves
  const size_t nW  = (size_t)3 * HIDv * HIDv;      // 4,915,200 halves
  const size_t nHS = (size_t)Bv * Hv * Sv * Dv;    // == nX
  _Float16* Xh = (_Float16*)ws;
  _Float16* Wp = (_Float16*)(ws + nX * 2);
  _Float16* Qh = (_Float16*)(ws + nX * 2 + nW * 2);
  _Float16* Kh = Qh + nHS;
  _Float16* Vt = Kh + nHS;

  cast_x_kernel<<<(int)((nX + 255) / 256), 256, 0, stream>>>(X, Xh, (int)nX);
  pack_w_kernel<<<(int)((nW + 255) / 256), 256, 0, stream>>>(Wq, Wk, Wv, Wp);

  // 256 row-tiles * 20 heads * 3 matrices = 15360 waves; 4 waves/block
  qkv_rope_kernel<<<15360 / 4, 128, 0, stream>>>(Xh, Wp, bq, bk, bv, Qh, Kh, Vt);

  // B*H*(S/64) = 1280 blocks, 4 waves each
  flash_attn_kernel<<<Bv * Hv * (Sv / 64), 128, 0, stream>>>(Qh, Kh, Vt, mask,
                                                             out);
}
